// LSTM_encoder_29652454211585
// MI455X (gfx1250) — matrix operand order repaired and checked by reading, not compiled, not run on hardware
//
#include <hip/hip_runtime.h>

#define TSTEPS 1024
#define NB 64          // batch
#define HID 384        // hidden / input dim
#define NCOL4 1536     // 4*HID
#define NWG 24         // 384 / 16 column blocks
#define THREADS 256    // 8 waves

typedef __attribute__((ext_vector_type(16))) _Float16 v16h;
typedef __attribute__((ext_vector_type(8)))  _Float16 v8h;
typedef __attribute__((ext_vector_type(8)))  float    v8f;

__device__ __forceinline__ float sigm_(float x) {
  return 1.0f / (1.0f + __expf(-x));
}
__device__ __forceinline__ float tanh_(float x) {
  float a = __expf(2.0f * fabsf(x));
  float r = 1.0f - 2.0f / (a + 1.0f);
  return copysignf(r, x);
}

// A-fragment for v_wmma_f32_16x16x32_f16 from a row-major [rows][384] f16 matrix.
// Lane L: row = arow, K = kk + 16*half + [0..15].
__device__ __forceinline__ v16h load_a_frag(const _Float16* __restrict__ base,
                                            int row, int kk, int hf) {
  const _Float16* p = base + row * HID + kk + 16 * hf;
  v8h lo = *(const v8h*)(p);
  v8h hi = *(const v8h*)(p + 8);
  v16h a;
#pragma unroll
  for (int i = 0; i < 8; ++i) { a[i] = lo[i]; a[8 + i] = hi[i]; }
  return a;
}

__global__ void __launch_bounds__(THREADS)
lstm_layer_kernel(const _Float16* __restrict__ xin,   // (T, B, 384) f16
                  const float* __restrict__ Wi,       // (384, 1536) f32
                  const float* __restrict__ Wh,       // (384, 1536) f32
                  const float* __restrict__ bias,     // (1536) f32
                  _Float16* __restrict__ xout,        // (T, B, 384) f16
                  float* __restrict__ fout,           // (B, T, 384) f32 or null
                  _Float16* __restrict__ hbuf,        // 2 * 64 * 384 f16 (double buffer)
                  unsigned* __restrict__ counter,     // zeroed before launch
                  int rev)
{
  // Per-WG weight slices, transposed to [col(64)][K(384)] so that every WMMA
  // B-fragment is a contiguous 32B chunk per lane (-> ds_load_b128 pairs).
  __shared__ _Float16 wiLds[64 * HID];   // 48 KB
  __shared__ _Float16 whLds[64 * HID];   // 48 KB
  __shared__ float    zbuf[4 * NB * 16]; // 16 KB gate pre-activations
  __shared__ float    cbuf[NB * 16];     //  4 KB cell state
  __shared__ float    bLds[64];

  const int tid  = threadIdx.x;
  const int jb   = blockIdx.x;      // which 16-wide column block of H
  const int wave = tid >> 5;
  const int lane = tid & 31;
  const int mblk = wave & 3;        // M block (16 batch rows)
  const int g0   = (wave >> 2) * 2; // gate pair {0,1} or {2,3}
  const int g1   = g0 + 1;
  const int hf   = lane >> 4;
  const int ln   = lane & 15;

  // Stage weight slices (f32 -> f16, transposed) into LDS. One-time per layer.
  for (int e = tid; e < 64 * HID; e += THREADS) {
    int c = e & 63;                                  // local col 0..63
    int k = e >> 6;                                  // 0..383
    int gcol = (c >> 4) * HID + jb * 16 + (c & 15);  // global gate column
    wiLds[c * HID + k] = (_Float16)Wi[k * NCOL4 + gcol];
    whLds[c * HID + k] = (_Float16)Wh[k * NCOL4 + gcol];
  }
  if (tid < 64) bLds[tid] = bias[(tid >> 4) * HID + jb * 16 + (tid & 15)];
  for (int e = tid; e < NB * 16; e += THREADS) cbuf[e] = 0.0f;
  __syncthreads();

  const int arow = mblk * 16 + ln;

  for (int t = 0; t < TSTEPS; ++t) {
    const int tt = rev ? (TSTEPS - 1 - t) : t;
    const _Float16* xt    = xin  + (size_t)tt * NB * HID;
    const _Float16* hprev = hbuf + (size_t)(t & 1) * (NB * HID);
    _Float16*       hnext = hbuf + (size_t)((t + 1) & 1) * (NB * HID);

    v8f acc0 = 0.0f;
    v8f acc1 = 0.0f;

    // z = x_t @ Wi  (K = 384, 12 k-steps, 2 gate tiles per wave)
#pragma unroll
    for (int kk = 0; kk < HID; kk += 32) {
      v16h a  = load_a_frag(xt, arow, kk, hf);
      v16h b0 = *(const v16h*)(wiLds + (g0 * 16 + ln) * HID + kk + 16 * hf);
      v16h b1 = *(const v16h*)(wiLds + (g1 * 16 + ln) * HID + kk + 16 * hf);
      acc0 = __builtin_amdgcn_wmma_f32_16x16x32_f16(false, a, false, b0, (short)0, acc0, false, false);
      acc1 = __builtin_amdgcn_wmma_f32_16x16x32_f16(false, a, false, b1, (short)0, acc1, false, false);
    }
    // z += h_{t-1} @ Wh   (h_0 == 0, so skip at t==0; branch is wave-uniform)
    if (t > 0) {
#pragma unroll
      for (int kk = 0; kk < HID; kk += 32) {
        v16h a  = load_a_frag(hprev, arow, kk, hf);
        v16h b0 = *(const v16h*)(whLds + (g0 * 16 + ln) * HID + kk + 16 * hf);
        v16h b1 = *(const v16h*)(whLds + (g1 * 16 + ln) * HID + kk + 16 * hf);
        acc0 = __builtin_amdgcn_wmma_f32_16x16x32_f16(false, a, false, b0, (short)0, acc0, false, false);
        acc1 = __builtin_amdgcn_wmma_f32_16x16x32_f16(false, a, false, b1, (short)0, acc1, false, false);
      }
    }

    // Scatter the two z tiles to LDS. C/D layout: VGPR v -> row M = v + 8*half, col = lane%16.
#pragma unroll
    for (int v = 0; v < 8; ++v) {
      int row = mblk * 16 + v + 8 * hf;
      zbuf[(g0 * NB + row) * 16 + ln] = acc0[v];
      zbuf[(g1 * NB + row) * 16 + ln] = acc1[v];
    }
    __syncthreads();

    // Elementwise gates: 1024 (row,col) cells, 4 per thread.
#pragma unroll
    for (int i = 0; i < 4; ++i) {
      int e    = tid + i * THREADS;
      int row  = e >> 4;
      int cc   = e & 15;
      int hcol = jb * 16 + cc;
      float zi = zbuf[(0 * NB + row) * 16 + cc] + bLds[cc];
      float zf = zbuf[(1 * NB + row) * 16 + cc] + bLds[16 + cc];
      float zg = zbuf[(2 * NB + row) * 16 + cc] + bLds[32 + cc];
      float zo = zbuf[(3 * NB + row) * 16 + cc] + bLds[48 + cc];
      float cnew = sigm_(zf) * cbuf[e] + sigm_(zi) * tanh_(zg);
      cbuf[e] = cnew;
      float h = sigm_(zo) * tanh_(cnew);
      hnext[row * HID + hcol] = (_Float16)h;
      xout[((size_t)tt * NB + row) * HID + hcol] = (_Float16)h;
      if (fout) fout[((size_t)row * TSTEPS + tt) * HID + hcol] = h;
    }

    // Device-wide barrier between time steps (h must be globally visible).
    __threadfence();
    __syncthreads();
    if (tid == 0) {
      __hip_atomic_fetch_add(counter, 1u, __ATOMIC_RELEASE, __HIP_MEMORY_SCOPE_AGENT);
      const unsigned target = gridDim.x * (unsigned)(t + 1);
      while (__hip_atomic_load(counter, __ATOMIC_ACQUIRE, __HIP_MEMORY_SCOPE_AGENT) < target) {
        __builtin_amdgcn_s_sleep(2);
      }
    }
    __syncthreads();
    __builtin_amdgcn_fence(__ATOMIC_ACQUIRE, "agent");
  }
}

// (B,T,D) f32  ->  (T,B,D) f16
__global__ void cvt_x_kernel(const float* __restrict__ xin, _Float16* __restrict__ xout) {
  const size_t total = (size_t)NB * TSTEPS * HID;
  for (size_t idx = blockIdx.x * (size_t)blockDim.x + threadIdx.x; idx < total;
       idx += (size_t)gridDim.x * blockDim.x) {
    size_t d = idx % HID;
    size_t t = (idx / HID) % TSTEPS;
    size_t b = idx / ((size_t)HID * TSTEPS);
    xout[(t * NB + b) * HID + d] = (_Float16)xin[idx];
  }
}

__global__ void init_kernel(unsigned* counters) {
  if (threadIdx.x < 8) counters[threadIdx.x] = 0u;
}

extern "C" void kernel_launch(void* const* d_in, const int* in_sizes, int n_in,
                              void* d_out, int out_size, void* d_ws, size_t ws_size,
                              hipStream_t stream) {
  (void)in_sizes; (void)n_in; (void)out_size; (void)ws_size;
  const float* x  = (const float*)d_in[0];
  const float* Wi = (const float*)d_in[1];
  const float* Wh = (const float*)d_in[2];
  const float* b  = (const float*)d_in[3];

  char* ws = (char*)d_ws;
  const size_t xbufBytes = (size_t)TSTEPS * NB * HID * sizeof(_Float16);  // ~48 MB
  _Float16* xA   = (_Float16*)(ws);
  _Float16* xB   = (_Float16*)(ws + xbufBytes);
  _Float16* hbuf = (_Float16*)(ws + 2 * xbufBytes);
  unsigned* ctrs = (unsigned*)(ws + 2 * xbufBytes +
                               2 * (size_t)NB * HID * sizeof(_Float16));

  hipLaunchKernelGGL(init_kernel, dim3(1), dim3(32), 0, stream, ctrs);
  hipLaunchKernelGGL(cvt_x_kernel, dim3(1024), dim3(256), 0, stream, x, xA);

  const int revs[5] = {1, 0, 1, 0, 1};
  _Float16* cur = xA;
  _Float16* nxt = xB;
  for (int l = 0; l < 5; ++l) {
    hipLaunchKernelGGL(lstm_layer_kernel, dim3(NWG), dim3(THREADS), 0, stream,
                       cur,
                       Wi + (size_t)l * HID * NCOL4,
                       Wh + (size_t)l * HID * NCOL4,
                       b  + (size_t)l * NCOL4,
                       nxt,
                       (l == 4) ? (float*)d_out : (float*)nullptr,
                       hbuf, ctrs + l, revs[l]);
    _Float16* tmp = cur; cur = nxt; nxt = tmp;
  }
}
